// AWD_LSTM_22531398435120
// MI455X (gfx1250) — matrix operand order repaired and checked by reading, not compiled, not run on hardware
//
#include <hip/hip_runtime.h>
#include <hip/hip_bf16.h>
#include <math.h>

// Problem sizes (fixed by the reference)
#define T_LEN 256
#define B_SZ  64
#define V_SZ  10000
#define E_SZ  400
#define E_P   416      // E padded to multiple of 32 (WMMA K granularity)
#define H_SZ  1152     // already a multiple of 32

typedef __bf16 v16bf __attribute__((ext_vector_type(16)));
typedef float  v8f   __attribute__((ext_vector_type(8)));
typedef unsigned int u32x4 __attribute__((ext_vector_type(4)));

union BF16Frag { v16bf v; unsigned short u[16]; unsigned int d[8]; u32x4 q[2]; };
union AccFrag  { v8f v; float f[8]; };

// float -> bf16 round-to-nearest-even (bit-level)
__device__ __forceinline__ unsigned short f2bf(float f) {
    unsigned int x = __float_as_uint(f);
    x = x + 0x7FFFu + ((x >> 16) & 1u);
    return (unsigned short)(x >> 16);
}
__device__ __forceinline__ float sigmoid_f(float x) {
    return 1.0f / (1.0f + __expf(-x));
}

// ---------------- WMMA fragment builders (16x16x32 bf16) ----------------
// All K extents are multiples of 32 and all rows 16B-aligned -> unguarded
// contiguous 16-byte vector loads (global_load_b128), zero conversion VALU.

// A (16x32, MxK): lane l: row m = mbase + (l&15); K runs {kh..kh+7} and {kh+16..kh+23}, kh = 8*(l>>4).
__device__ __forceinline__ BF16Frag frag_a_bf16(const unsigned short* __restrict__ X, int ld,
                                                int mbase, int kbase, int lane) {
    const unsigned short* p = X + (size_t)(mbase + (lane & 15)) * (size_t)ld + kbase + ((lane >> 4) * 8);
    BF16Frag fr;
    fr.q[0] = *(const u32x4*)(p);        // k offsets 0..7
    fr.q[1] = *(const u32x4*)(p + 16);   // k offsets 16..23
    return fr;
}

// B (32x16, KxN), weights row-major W[n][k] (x @ W^T):
// lane l: n = nbase + (l&15); K runs {kh..kh+7} and {kh+16..kh+23}, kh = 8*(l>>4).
__device__ __forceinline__ BF16Frag frag_b_bf16(const unsigned short* __restrict__ W, int ld,
                                                int nbase, int kbase, int lane) {
    const unsigned short* p = W + (size_t)(nbase + (lane & 15)) * (size_t)ld + kbase + ((lane >> 4) * 8);
    BF16Frag fr;
    fr.q[0] = *(const u32x4*)(p);
    fr.q[1] = *(const u32x4*)(p + 16);
    return fr;
}

__device__ __forceinline__ v8f wmma_bf16(const BF16Frag& a, const BF16Frag& b, v8f acc) {
    return __builtin_amdgcn_wmma_f32_16x16x32_bf16(
        /*neg_a=*/false, a.v, /*neg_b=*/false, b.v,
        /*c_mod=*/(short)0, acc, /*reuse_a=*/false, /*reuse_b=*/false);
}

// ---------------- Utility kernels ----------------
// f32[rows][K] -> bf16[rows][Kp], zero-padding columns K..Kp-1.
__global__ __launch_bounds__(256) void awd_cvt_pad_kernel(const float* __restrict__ src,
                                                          unsigned short* __restrict__ dst,
                                                          int rows, int K, int Kp) {
    int i = blockIdx.x * blockDim.x + threadIdx.x;
    int n = rows * Kp;
    int stride = gridDim.x * blockDim.x;
    for (; i < n; i += stride) {
        int r = i / Kp;
        int k = i - r * Kp;
        dst[i] = (k < K) ? f2bf(src[(size_t)r * K + k]) : (unsigned short)0;
    }
}

__global__ __launch_bounds__(256) void awd_zero_u16_kernel(unsigned short* __restrict__ p, int n) {
    int i = blockIdx.x * blockDim.x + threadIdx.x;
    int stride = gridDim.x * blockDim.x;
    for (; i < n; i += stride) p[i] = 0;
}

__global__ __launch_bounds__(256) void awd_bias_sum_kernel(const float* __restrict__ a,
                                                           const float* __restrict__ b,
                                                           float* __restrict__ o, int n) {
    int i = blockIdx.x * blockDim.x + threadIdx.x;
    if (i < n) o[i] = a[i] + b[i];
}

// Embedding gather directly to padded bf16 rows: xe[tb][E_P].
__global__ __launch_bounds__(256) void awd_embed_kernel(const int* __restrict__ x,
                                                        const float* __restrict__ emb,
                                                        unsigned short* __restrict__ xe, int tb_total) {
    int i = blockIdx.x * blockDim.x + threadIdx.x;
    int n = tb_total * E_P;
    int stride = gridDim.x * blockDim.x;
    for (; i < n; i += stride) {
        int tb = i / E_P;
        int e  = i - tb * E_P;
        xe[i] = (e < E_SZ) ? f2bf(emb[(size_t)x[tb] * E_SZ + e]) : (unsigned short)0;
    }
}

// ---------------- Fused LSTM layer step ----------------
// gates = x_t @ Wih^T + h_prev @ Whh^T + (b_ih + b_hh); gate order i,f,g,o.
// One wave per block computes ALL 4 batch tiles (B=64) for one 16-wide dout
// column: per K-chunk it loads 4 A-fragments + 1 B-fragment per gate and
// issues 16 WMMAs, so each weight row is fetched exactly once per step.
// 16 f32 accumulators = 128 VGPRs; __launch_bounds__(32,1) opens the full
// register window so they stay resident (no scratch spills).
// grid = (dout/16), block = 32.
__global__ __launch_bounds__(32, 1)
void awd_lstm_step_kernel(const unsigned short* __restrict__ xt, int x_ld,     // [B][x_ld] bf16
                          const unsigned short* __restrict__ h_prev, int h_ld, // [B][h_ld] bf16
                          const float* __restrict__ c_prev,                    // [B][dout] f32
                          const unsigned short* __restrict__ Wih,   // [4*dout][x_ld]
                          const unsigned short* __restrict__ Whh,   // [4*dout][h_ld]
                          const float* __restrict__ biasSum,        // [4*dout]
                          unsigned short* __restrict__ h_out,       // [B][h_ld] bf16
                          float* __restrict__ c_out,                // [B][dout] f32
                          int dout) {
    const int lane  = threadIdx.x & 31;
    const int ntile = blockIdx.x * 16;   // column within dout

    AccFrag acc[4][4];                   // [gate][batch-tile]
#pragma unroll
    for (int g = 0; g < 4; ++g)
#pragma unroll
        for (int m = 0; m < 4; ++m)
#pragma unroll
            for (int s = 0; s < 8; ++s) acc[g][m].f[s] = 0.0f;

    // x_t contribution
    for (int kb = 0; kb < x_ld; kb += 32) {
        BF16Frag a0 = frag_a_bf16(xt, x_ld,  0, kb, lane);
        BF16Frag a1 = frag_a_bf16(xt, x_ld, 16, kb, lane);
        BF16Frag a2 = frag_a_bf16(xt, x_ld, 32, kb, lane);
        BF16Frag a3 = frag_a_bf16(xt, x_ld, 48, kb, lane);
#pragma unroll
        for (int g = 0; g < 4; ++g) {
            BF16Frag b = frag_b_bf16(Wih, x_ld, g * dout + ntile, kb, lane);
            acc[g][0].v = wmma_bf16(a0, b, acc[g][0].v);
            acc[g][1].v = wmma_bf16(a1, b, acc[g][1].v);
            acc[g][2].v = wmma_bf16(a2, b, acc[g][2].v);
            acc[g][3].v = wmma_bf16(a3, b, acc[g][3].v);
        }
    }
    // h_{t-1} contribution
    for (int kb = 0; kb < h_ld; kb += 32) {
        BF16Frag a0 = frag_a_bf16(h_prev, h_ld,  0, kb, lane);
        BF16Frag a1 = frag_a_bf16(h_prev, h_ld, 16, kb, lane);
        BF16Frag a2 = frag_a_bf16(h_prev, h_ld, 32, kb, lane);
        BF16Frag a3 = frag_a_bf16(h_prev, h_ld, 48, kb, lane);
#pragma unroll
        for (int g = 0; g < 4; ++g) {
            BF16Frag b = frag_b_bf16(Whh, h_ld, g * dout + ntile, kb, lane);
            acc[g][0].v = wmma_bf16(a0, b, acc[g][0].v);
            acc[g][1].v = wmma_bf16(a1, b, acc[g][1].v);
            acc[g][2].v = wmma_bf16(a2, b, acc[g][2].v);
            acc[g][3].v = wmma_bf16(a3, b, acc[g][3].v);
        }
    }

    // Elementwise cell update: lane l slot s -> (m = s + 8*(l>>4), n = l&15)
    const int nn   = ntile + (lane & 15);
    const int half = lane >> 4;
    const float bi = biasSum[0 * dout + nn];
    const float bf = biasSum[1 * dout + nn];
    const float bg = biasSum[2 * dout + nn];
    const float bo = biasSum[3 * dout + nn];

#pragma unroll
    for (int mi = 0; mi < 4; ++mi) {
#pragma unroll
        for (int s = 0; s < 8; ++s) {
            int bm = mi * 16 + s + 8 * half;
            float gi = sigmoid_f(acc[0][mi].f[s] + bi);
            float gf = sigmoid_f(acc[1][mi].f[s] + bf);
            float gg = tanhf(acc[2][mi].f[s] + bg);
            float go = sigmoid_f(acc[3][mi].f[s] + bo);
            float cn = gf * c_prev[(size_t)bm * dout + nn] + gi * gg;
            float hn = go * tanhf(cn);
            c_out[(size_t)bm * dout + nn] = cn;
            h_out[(size_t)bm * h_ld + nn] = f2bf(hn);
        }
    }
}

// ---------------- Decoder GEMM: C[M,N] = A[M,Kp] @ Bw[N,Kp]^T + bias[N] ----------------
// Register-blocked 2Mx5N tiles per wave (N tiles: 625 = 125*5, exact fit).
// block = 128 (4 waves), grid = (125, M/(32*4) = 128).
__global__ __launch_bounds__(128, 1)
void awd_decoder_gemm_kernel(const unsigned short* __restrict__ A,
                             const unsigned short* __restrict__ Bw,
                             const float* __restrict__ bias,
                             float* __restrict__ C,
                             int N, int Kp) {
    const int lane  = threadIdx.x & 31;
    const int wave  = threadIdx.x >> 5;
    const int nbase = blockIdx.x * 80;                 // 5 N-tiles
    const int mbase = (blockIdx.y * 4 + wave) * 32;    // 2 M-tiles

    AccFrag acc[2][5];
#pragma unroll
    for (int mi = 0; mi < 2; ++mi)
#pragma unroll
        for (int j = 0; j < 5; ++j)
#pragma unroll
            for (int s = 0; s < 8; ++s) acc[mi][j].f[s] = 0.0f;

    for (int kb = 0; kb < Kp; kb += 32) {
        BF16Frag a0 = frag_a_bf16(A, Kp, mbase,      kb, lane);
        BF16Frag a1 = frag_a_bf16(A, Kp, mbase + 16, kb, lane);
#pragma unroll
        for (int j = 0; j < 5; ++j) {
            BF16Frag b = frag_b_bf16(Bw, Kp, nbase + j * 16, kb, lane);
            acc[0][j].v = wmma_bf16(a0, b, acc[0][j].v);
            acc[1][j].v = wmma_bf16(a1, b, acc[1][j].v);
        }
    }

    const int half = lane >> 4;
#pragma unroll
    for (int j = 0; j < 5; ++j) {
        int nn = nbase + j * 16 + (lane & 15);
        float bn = bias[nn];
#pragma unroll
        for (int mi = 0; mi < 2; ++mi)
#pragma unroll
            for (int s = 0; s < 8; ++s) {
                int m = mbase + mi * 16 + s + 8 * half;
                C[(size_t)m * N + nn] = acc[mi][j].f[s] + bn;
            }
    }
}

// ---------------- Host launcher ----------------
extern "C" void kernel_launch(void* const* d_in, const int* in_sizes, int n_in,
                              void* d_out, int out_size, void* d_ws, size_t ws_size,
                              hipStream_t stream) {
    (void)in_sizes; (void)n_in; (void)out_size; (void)ws_size;

    const int*   x     = (const int*)  d_in[0];
    const float* h0_in = (const float*)d_in[1];
    const float* h1_in = (const float*)d_in[2];
    const float* h2_in = (const float*)d_in[3];
    const float* c0_in = (const float*)d_in[4];
    const float* c1_in = (const float*)d_in[5];
    const float* c2_in = (const float*)d_in[6];
    const float* emb   = (const float*)d_in[7];
    const float* W_ih0 = (const float*)d_in[8];
    const float* W_hh0 = (const float*)d_in[9];
    const float* b_ih0 = (const float*)d_in[10];
    const float* b_hh0 = (const float*)d_in[11];
    const float* W_ih1 = (const float*)d_in[12];
    const float* W_hh1 = (const float*)d_in[13];
    const float* b_ih1 = (const float*)d_in[14];
    const float* b_hh1 = (const float*)d_in[15];
    const float* W_ih2 = (const float*)d_in[16];
    const float* W_hh2 = (const float*)d_in[17];
    const float* b_ih2 = (const float*)d_in[18];
    const float* b_hh2 = (const float*)d_in[19];
    const float* dec_W = (const float*)d_in[20];
    const float* dec_b = (const float*)d_in[21];
    float* out = (float*)d_out;

    // ---- carve workspace (d_ws assumed 256B-aligned) ----
    char* ws = (char*)d_ws;
    size_t off = 0;
    auto carve = [&](size_t bytes) -> void* {
        void* p = ws + off;
        off = (off + bytes + 255) & ~(size_t)255;
        return p;
    };

    const size_t TB = (size_t)T_LEN * B_SZ;
    unsigned short* xe   = (unsigned short*)carve(TB * E_P * 2);   // bf16 embedded inputs
    unsigned short* ybf  = (unsigned short*)carve(TB * E_P * 2);   // bf16 layer-2 outputs (= h2 history)
    unsigned short* wih0 = (unsigned short*)carve((size_t)4 * H_SZ * E_P  * 2);
    unsigned short* whh0 = (unsigned short*)carve((size_t)4 * H_SZ * H_SZ * 2);
    unsigned short* wih1 = (unsigned short*)carve((size_t)4 * H_SZ * H_SZ * 2);
    unsigned short* whh1 = (unsigned short*)carve((size_t)4 * H_SZ * H_SZ * 2);
    unsigned short* wih2 = (unsigned short*)carve((size_t)4 * E_SZ * H_SZ * 2);
    unsigned short* whh2 = (unsigned short*)carve((size_t)4 * E_SZ * E_P  * 2);
    unsigned short* wdec = (unsigned short*)carve((size_t)V_SZ * E_P * 2);
    float* bs0 = (float*)carve(4 * H_SZ * sizeof(float));
    float* bs1 = (float*)carve(4 * H_SZ * sizeof(float));
    float* bs2 = (float*)carve(4 * E_SZ * sizeof(float));
    // bf16 h state (ping-pong for layers 0/1; layer 2 uses ybf rows as history)
    unsigned short* h0b[2]; unsigned short* h1b[2];
    unsigned short* h2init = (unsigned short*)carve((size_t)B_SZ * E_P * 2);
    float *c0b[2], *c1b[2], *c2b[2];
    for (int p = 0; p < 2; ++p) {
        h0b[p] = (unsigned short*)carve((size_t)B_SZ * H_SZ * 2);
        h1b[p] = (unsigned short*)carve((size_t)B_SZ * H_SZ * 2);
        c0b[p] = (float*)carve((size_t)B_SZ * H_SZ * sizeof(float));
        c1b[p] = (float*)carve((size_t)B_SZ * H_SZ * sizeof(float));
        c2b[p] = (float*)carve((size_t)B_SZ * E_SZ * sizeof(float));
    }

    auto gsz = [](int n) { int b = (n + 255) / 256; return b > 16384 ? 16384 : b; };

    // ---- initial recurrent state ----
    // h -> bf16 (padded where needed); c -> f32 contiguous copies
    awd_cvt_pad_kernel<<<gsz(B_SZ * H_SZ), 256, 0, stream>>>(h0_in, h0b[0], B_SZ, H_SZ, H_SZ);
    awd_cvt_pad_kernel<<<gsz(B_SZ * H_SZ), 256, 0, stream>>>(h1_in, h1b[0], B_SZ, H_SZ, H_SZ);
    awd_cvt_pad_kernel<<<gsz(B_SZ * E_P),  256, 0, stream>>>(h2_in, h2init, B_SZ, E_SZ, E_P);
    hipMemcpyAsync(c0b[0], c0_in, (size_t)B_SZ * H_SZ * 4, hipMemcpyDeviceToDevice, stream);
    hipMemcpyAsync(c1b[0], c1_in, (size_t)B_SZ * H_SZ * 4, hipMemcpyDeviceToDevice, stream);
    hipMemcpyAsync(c2b[0], c2_in, (size_t)B_SZ * E_SZ * 4, hipMemcpyDeviceToDevice, stream);

    // ---- weights -> bf16 (K padded where needed) ----
    awd_cvt_pad_kernel<<<gsz(4 * H_SZ * E_P),  256, 0, stream>>>(W_ih0, wih0, 4 * H_SZ, E_SZ, E_P);
    awd_cvt_pad_kernel<<<gsz(4 * H_SZ * H_SZ), 256, 0, stream>>>(W_hh0, whh0, 4 * H_SZ, H_SZ, H_SZ);
    awd_cvt_pad_kernel<<<gsz(4 * H_SZ * H_SZ), 256, 0, stream>>>(W_ih1, wih1, 4 * H_SZ, H_SZ, H_SZ);
    awd_cvt_pad_kernel<<<gsz(4 * H_SZ * H_SZ), 256, 0, stream>>>(W_hh1, whh1, 4 * H_SZ, H_SZ, H_SZ);
    awd_cvt_pad_kernel<<<gsz(4 * E_SZ * H_SZ), 256, 0, stream>>>(W_ih2, wih2, 4 * E_SZ, H_SZ, H_SZ);
    awd_cvt_pad_kernel<<<gsz(4 * E_SZ * E_P),  256, 0, stream>>>(W_hh2, whh2, 4 * E_SZ, E_SZ, E_P);
    awd_cvt_pad_kernel<<<gsz(V_SZ * E_P),      256, 0, stream>>>(dec_W, wdec, V_SZ, E_SZ, E_P);

    awd_bias_sum_kernel<<<(4 * H_SZ + 255) / 256, 256, 0, stream>>>(b_ih0, b_hh0, bs0, 4 * H_SZ);
    awd_bias_sum_kernel<<<(4 * H_SZ + 255) / 256, 256, 0, stream>>>(b_ih1, b_hh1, bs1, 4 * H_SZ);
    awd_bias_sum_kernel<<<(4 * E_SZ + 255) / 256, 256, 0, stream>>>(b_ih2, b_hh2, bs2, 4 * E_SZ);

    // ---- zero ybf: pad columns are read as A-fragments (layer-2 recurrence + decoder) ----
    awd_zero_u16_kernel<<<gsz((int)(TB * E_P)), 256, 0, stream>>>(ybf, (int)(TB * E_P));

    // ---- embedding gather (bf16, padded rows) ----
    awd_embed_kernel<<<gsz((int)(TB * E_P)), 256, 0, stream>>>(x, emb, xe, (int)TB);

    // ---- sequential LSTM scan: 3 fused layer-step kernels per timestep ----
    const dim3 gridH(H_SZ / 16);   // 72 blocks x 1 wave
    const dim3 gridE(E_SZ / 16);   // 25 blocks x 1 wave
    for (int t = 0; t < T_LEN; ++t) {
        const int cur = t & 1, nxt = cur ^ 1;
        // layer 0: E -> H
        awd_lstm_step_kernel<<<gridH, 32, 0, stream>>>(
            xe + (size_t)t * B_SZ * E_P, E_P,
            h0b[cur], H_SZ, c0b[cur],
            wih0, whh0, bs0, h0b[nxt], c0b[nxt], H_SZ);
        // layer 1: H -> H
        awd_lstm_step_kernel<<<gridH, 32, 0, stream>>>(
            h0b[nxt], H_SZ,
            h1b[cur], H_SZ, c1b[cur],
            wih1, whh1, bs1, h1b[nxt], c1b[nxt], H_SZ);
        // layer 2: H -> E; h state IS the output history in ybf
        const unsigned short* h2prev = (t == 0) ? h2init : (ybf + (size_t)(t - 1) * B_SZ * E_P);
        awd_lstm_step_kernel<<<gridE, 32, 0, stream>>>(
            h1b[nxt], H_SZ,
            h2prev, E_P, c2b[cur],
            wih2, whh2, bs2,
            ybf + (size_t)t * B_SZ * E_P, c2b[nxt], E_SZ);
    }

    // ---- decoder: [T*B, E_P] @ dec_W^T + dec_b -> [T*B, V] ----
    {
        dim3 grid((V_SZ / 16) / 5, (int)(TB / 16 / 2 / 4));   // (125, 128)
        awd_decoder_gemm_kernel<<<grid, 128, 0, stream>>>(ybf, wdec, dec_b, out, V_SZ, E_P);
    }
}
